// EncoderNetwork_36146444763551
// MI455X (gfx1250) — matrix operand order repaired and checked by reading, not compiled, not run on hardware
//
#include <hip/hip_runtime.h>
#include <math.h>

// ---------------------------------------------------------------------------
// Types for CDNA5 WMMA (wave32): D[16x16 f32] = A[16x32 bf16] x B[32x16 bf16] + C
// ---------------------------------------------------------------------------
typedef __bf16 bf16;
typedef __attribute__((ext_vector_type(16))) __bf16 v16bf;
typedef __attribute__((ext_vector_type(8)))  float  v8f;

typedef unsigned int u32x4 __attribute__((ext_vector_type(4)));
typedef int          i32x4 __attribute__((ext_vector_type(4)));
typedef int          i32x8 __attribute__((ext_vector_type(8)));

union Frag16 { v16bf v; uint4 u[2]; bf16 h[16]; };
union B4     { uint2 u; bf16 h[4]; };

#define T_SEQ  512
#define DMODEL 1024
#define NHEAD  16
#define HSZ    64
#define DFF    10240
#define NB     8
#define RELBLK 512

#if __has_builtin(__builtin_amdgcn_tensor_load_to_lds) && \
    __has_builtin(__builtin_amdgcn_s_wait_tensorcnt)
#define USE_TDM 1
#else
#define USE_TDM 0
#endif

static __device__ inline v8f wmma_bf16(const Frag16& a, const Frag16& b, v8f c)
{
    return __builtin_amdgcn_wmma_f32_16x16x32_bf16(false, a.v, false, b.v,
                                                   (short)0, c, false, false);
}

#if USE_TDM
// ---------------------------------------------------------------------------
// TDM: DMA a [128 rows x 64 elems] bf16 panel (row stride ldb elems) from
// global into LDS at byte offset lds_addr, packed rows (row stride 64 elems =
// 128 B) -- exactly the ldsB layout the fragment gathers expect.
// D# per ISA ch.8: group0 {count=1, lds_addr, global_addr[56:0], type=2},
// group1 {data_size=1(2B), tensor_dim0=64, tensor_dim1=128, tile_dim0=64,
// tile_dim1=128, tensor_dim0_stride=ldb}. Groups 2/3 zero (2-D tensor).
// ---------------------------------------------------------------------------
static __device__ inline void tdm_load_b_panel(const bf16* gptr, int ldb,
                                               unsigned lds_addr)
{
    const unsigned long long ga = (unsigned long long)(size_t)gptr;
    u32x4 g0;
    g0.x = 1u;                                   // count=1, is_restore=0
    g0.y = lds_addr;                             // lds_addr (bytes)
    g0.z = (unsigned)(ga & 0xffffffffu);         // global_addr[31:0]
    g0.w = (unsigned)((ga >> 32) & 0x01ffffffu)  // global_addr[56:32]
         | (2u << 30);                           // type=2 ("image")
    i32x8 g1;
    g1[0] = 0x00010000;          // wg_mask=0, data_size=1 (2 bytes), flags=0
    g1[1] = 64 << 16;            // tensor_dim0[15:0]=64 (bits 63:48)
    g1[2] = 128 << 16;           // tensor_dim0[31:16]=0 | tensor_dim1[15:0]=128
    g1[3] = 64 << 16;            // tensor_dim1[31:16]=0 | tile_dim0=64
    g1[4] = 128;                 // tile_dim1=128, tile_dim2=0
    g1[5] = ldb;                 // tensor_dim0_stride[31:0] (elements)
    g1[6] = 0;                   // stride[47:32]=0, dim1_stride[15:0]=0
    g1[7] = 0;
    const i32x4 gz = {0, 0, 0, 0};
#if defined(__clang_major__) && (__clang_major__ >= 23)
    const i32x8 gz8 = {0, 0, 0, 0, 0, 0, 0, 0};
    __builtin_amdgcn_tensor_load_to_lds(g0, g1, gz, gz, gz8, 0);
#else
    __builtin_amdgcn_tensor_load_to_lds(g0, g1, gz, gz, 0);
#endif
}
#endif  // USE_TDM

// ---------------------------------------------------------------------------
// Fragment layouts used here (16-bit A 16x32 / B 32x16; contiguous k, not the ISA 7.12.2 table order):
//   A frag, lane L: row = L&15, kb = (L<16)?0:16 -> elems {kb..kb+15}
//   B frag, lane L: col = L&15, ko = (L<16)?0:16 -> elems {ko..ko+15}
// All B operands are stored [N,K] row-major (pre-transposed), so LDS staging is
// pure contiguous B128 copies and fragment gathers are contiguous ds_load_b128.
// ---------------------------------------------------------------------------

// ============================ 64M x 128N core ==============================
// 8 waves in a 2(M) x 4(N) grid; each wave owns a 32x32 register tile. BK=64
// -> 8 WMMAs per wave per stage. LDS: A 64x64 (8KB) + B 128x64 (16KB).
template<bool A_F32>
static __device__ inline void gemm_core2(const void* Ap, int lda,
                                         const bf16* BT, int ldb, int K,
                                         bf16* ldsA, bf16* ldsB, v8f acc[4])
{
    const int tid  = threadIdx.x;
    const int lane = tid & 31;
    const int wave = tid >> 5;
    const int wm   = (wave >> 2) << 5;   // 0, 32
    const int wn   = (wave & 3) << 5;    // 0, 32, 64, 96

    for (int k0 = 0; k0 < K; k0 += 64) {
        {   // ---- stage A tile: 64 rows x 64 k ----
            const int r = tid >> 2;
            const int c = (tid & 3) << 4;          // 16 elems per thread
            if (A_F32) {
                const float4* s = (const float4*)((const float*)Ap + (size_t)r * lda + k0 + c);
                const float4 f0 = s[0], f1 = s[1], f2 = s[2], f3 = s[3];
                Frag16 t;
                t.h[0]=(bf16)f0.x; t.h[1]=(bf16)f0.y; t.h[2]=(bf16)f0.z; t.h[3]=(bf16)f0.w;
                t.h[4]=(bf16)f1.x; t.h[5]=(bf16)f1.y; t.h[6]=(bf16)f1.z; t.h[7]=(bf16)f1.w;
                t.h[8]=(bf16)f2.x; t.h[9]=(bf16)f2.y; t.h[10]=(bf16)f2.z; t.h[11]=(bf16)f2.w;
                t.h[12]=(bf16)f3.x; t.h[13]=(bf16)f3.y; t.h[14]=(bf16)f3.z; t.h[15]=(bf16)f3.w;
                *(uint4*)&ldsA[r * 64 + c]     = t.u[0];
                *(uint4*)&ldsA[r * 64 + c + 8] = t.u[1];
            } else {
                const bf16* s = (const bf16*)Ap + (size_t)r * lda + k0 + c;
                *(uint4*)&ldsA[r * 64 + c]     = *(const uint4*)s;
                *(uint4*)&ldsA[r * 64 + c + 8] = *(const uint4*)(s + 8);
            }
        }
        {   // ---- stage B tile: 128 n-rows x 64 k (contiguous copy) ----
            const int n = tid >> 1;
            const int c = (tid & 1) << 5;          // 32 elems per thread
            const bf16* s = BT + (size_t)n * ldb + k0 + c;
            *(uint4*)&ldsB[n * 64 + c]      = *(const uint4*)s;
            *(uint4*)&ldsB[n * 64 + c + 8]  = *(const uint4*)(s + 8);
            *(uint4*)&ldsB[n * 64 + c + 16] = *(const uint4*)(s + 16);
            *(uint4*)&ldsB[n * 64 + c + 24] = *(const uint4*)(s + 24);
            if (k0 + 64 < K)
                __builtin_prefetch(s + 64, 0, 1);
        }
        __syncthreads();

        #pragma unroll
        for (int kc = 0; kc < 2; ++kc) {
            const int kb = kc * 32 + ((lane < 16) ? 0 : 16);
            const int ko = kc * 32 + ((lane < 16) ? 0 : 16);
            const int ar = wm + (lane & 15);
            const int bn = wn + (lane & 15);
            Frag16 a0, a1, b0, b1;
            a0.u[0] = *(const uint4*)&ldsA[ar * 64 + kb];
            a0.u[1] = *(const uint4*)&ldsA[ar * 64 + kb + 8];
            a1.u[0] = *(const uint4*)&ldsA[(ar + 16) * 64 + kb];
            a1.u[1] = *(const uint4*)&ldsA[(ar + 16) * 64 + kb + 8];
            b0.u[0] = *(const uint4*)&ldsB[bn * 64 + ko];
            b0.u[1] = *(const uint4*)&ldsB[bn * 64 + ko + 8];
            b1.u[0] = *(const uint4*)&ldsB[(bn + 16) * 64 + ko];
            b1.u[1] = *(const uint4*)&ldsB[(bn + 16) * 64 + ko + 8];
            acc[0] = wmma_bf16(a0, b0, acc[0]);
            acc[1] = wmma_bf16(a0, b1, acc[1]);
            acc[2] = wmma_bf16(a1, b0, acc[2]);
            acc[3] = wmma_bf16(a1, b1, acc[3]);
        }
        __syncthreads();
    }
}

#if USE_TDM
// Same 64x128 core, but the B panel is DMA'd by the Tensor Data Mover with
// double buffering: wave 0 issues panel i+1's tensor_load_to_lds, waits
// s_wait_tensorcnt<=1 (=> panel i complete, in-order per wave), and the
// workgroup barrier publishes the LDS panel to the other 7 waves.
static __device__ inline void gemm_core2_tdm(const bf16* Ap, int lda,
                                             const bf16* BT, int ldb, int K,
                                             bf16* ldsA, bf16* ldsB0, bf16* ldsB1,
                                             v8f acc[4])
{
    const int tid  = threadIdx.x;
    const int lane = tid & 31;
    const int wave = tid >> 5;
    const int wm   = (wave >> 2) << 5;
    const int wn   = (wave & 3) << 5;
    const unsigned b0a = (unsigned)(size_t)ldsB0;   // LDS aperture: low 32 bits
    const unsigned b1a = (unsigned)(size_t)ldsB1;   //   = LDS byte offset
    const int nk = K >> 6;

    if (wave == 0)
        tdm_load_b_panel(BT, ldb, b0a);

    for (int i = 0; i < nk; ++i) {
        const int k0 = i << 6;
        const bf16* curB = (i & 1) ? ldsB1 : ldsB0;
        if (wave == 0) {
            if (i + 1 < nk) {
                tdm_load_b_panel(BT + (size_t)(k0 + 64), ldb, (i & 1) ? b0a : b1a);
                __builtin_amdgcn_s_wait_tensorcnt((short)1);   // panel i done
            } else {
                __builtin_amdgcn_s_wait_tensorcnt((short)0);
            }
        }
        {   // stage A tile 64x64 (small; stays on the manual path)
            const int r = tid >> 2;
            const int c = (tid & 3) << 4;
            const bf16* s = Ap + (size_t)r * lda + k0 + c;
            *(uint4*)&ldsA[r * 64 + c]     = *(const uint4*)s;
            *(uint4*)&ldsA[r * 64 + c + 8] = *(const uint4*)(s + 8);
        }
        __syncthreads();

        #pragma unroll
        for (int kc = 0; kc < 2; ++kc) {
            const int kb = kc * 32 + ((lane < 16) ? 0 : 16);
            const int ko = kc * 32 + ((lane < 16) ? 0 : 16);
            const int ar = wm + (lane & 15);
            const int bn = wn + (lane & 15);
            Frag16 a0, a1, b0, b1;
            a0.u[0] = *(const uint4*)&ldsA[ar * 64 + kb];
            a0.u[1] = *(const uint4*)&ldsA[ar * 64 + kb + 8];
            a1.u[0] = *(const uint4*)&ldsA[(ar + 16) * 64 + kb];
            a1.u[1] = *(const uint4*)&ldsA[(ar + 16) * 64 + kb + 8];
            b0.u[0] = *(const uint4*)&curB[bn * 64 + ko];
            b0.u[1] = *(const uint4*)&curB[bn * 64 + ko + 8];
            b1.u[0] = *(const uint4*)&curB[(bn + 16) * 64 + ko];
            b1.u[1] = *(const uint4*)&curB[(bn + 16) * 64 + ko + 8];
            acc[0] = wmma_bf16(a0, b0, acc[0]);
            acc[1] = wmma_bf16(a0, b1, acc[1]);
            acc[2] = wmma_bf16(a1, b0, acc[2]);
            acc[3] = wmma_bf16(a1, b1, acc[3]);
        }
        __syncthreads();
    }
}
#endif  // USE_TDM

#define EPI2_COORDS()                                               \
    const int lane = threadIdx.x & 31;                              \
    const int wave = threadIdx.x >> 5;                              \
    const int wm_  = ((wave >> 2) << 5) + ((lane < 16) ? 0 : 8);    \
    const int wn_  = ((wave & 3) << 5) + (lane & 15);

// ============================ 32M x 64N core ===============================
// For the hs=64-wide GEMMs (QKV projection, probs @ V^T). B is [N,K] only.
template<bool A_F32>
static __device__ inline v8f gemm_core_n64(const void* Ap, int lda,
                                           const bf16* BT, int ldb, int K,
                                           bf16* ldsA, bf16* ldsB)
{
    const int tid  = threadIdx.x;
    const int lane = tid & 31;
    const int wave = tid >> 5;
    const int mo   = (wave >> 2) << 4;   // 0 or 16
    const int no   = (wave & 3) << 4;    // 0,16,32,48
    v8f acc = {0.f, 0.f, 0.f, 0.f, 0.f, 0.f, 0.f, 0.f};

    for (int k0 = 0; k0 < K; k0 += 32) {
        {   // stage A: 32 rows x 32 k
            const int r = tid >> 3;
            const int c = (tid & 7) << 2;
            if (A_F32) {
                const float* s = (const float*)Ap + (size_t)r * lda + k0 + c;
                B4 t;
                t.h[0]=(bf16)s[0]; t.h[1]=(bf16)s[1]; t.h[2]=(bf16)s[2]; t.h[3]=(bf16)s[3];
                *(uint2*)&ldsA[r * 32 + c] = t.u;
            } else {
                const bf16* s = (const bf16*)Ap + (size_t)r * lda + k0 + c;
                *(uint2*)&ldsA[r * 32 + c] = *(const uint2*)s;
            }
        }
        {   // stage B: 64 n-rows x 32 k (contiguous copy)
            const int n = tid >> 2;
            const int c = (tid & 3) << 3;
            const bf16* s = BT + (size_t)n * ldb + k0 + c;
            *(uint4*)&ldsB[n * 32 + c] = *(const uint4*)s;
            if (k0 + 32 < K)
                __builtin_prefetch(s + 32, 0, 1);
        }
        __syncthreads();

        Frag16 a, b;
        const int ar = mo + (lane & 15);
        const int kb = (lane < 16) ? 0 : 16;
        a.u[0] = *(const uint4*)&ldsA[ar * 32 + kb];
        a.u[1] = *(const uint4*)&ldsA[ar * 32 + kb + 8];
        const int bn = no + (lane & 15);
        const int ko = (lane < 16) ? 0 : 16;
        b.u[0] = *(const uint4*)&ldsB[bn * 32 + ko];
        b.u[1] = *(const uint4*)&ldsB[bn * 32 + ko + 8];
        acc = wmma_bf16(a, b, acc);
        __syncthreads();
    }
    return acc;
}

#define EPI1_COORDS()                                               \
    const int lane = threadIdx.x & 31;                              \
    const int wave = threadIdx.x >> 5;                              \
    const int no_  = (wave & 3) << 4;                               \
    const int mo_  = ((wave >> 2) << 4) + ((lane < 16) ? 0 : 8);

// ---------------------------------------------------------------------------
// fp32 -> bf16 bulk conversion (n multiple of 4)
// ---------------------------------------------------------------------------
__global__ __launch_bounds__(256)
void cvt_f32_bf16_kernel(const float* __restrict__ in, bf16* __restrict__ out, long n4)
{
    const long i = (long)blockIdx.x * blockDim.x + threadIdx.x;
    if (i >= n4) return;
    const float4 f = ((const float4*)in)[i];
    B4 t; t.h[0] = (bf16)f.x; t.h[1] = (bf16)f.y; t.h[2] = (bf16)f.z; t.h[3] = (bf16)f.w;
    ((uint2*)out)[i] = t.u;
}

// ---------------------------------------------------------------------------
// fp32 [rows, cols] -> bf16 [cols, rows], batched. grid (cols/32, rows/32, batch),
// block (32, 8). LDS 32x33 tile transpose; both global phases are coalesced.
// ---------------------------------------------------------------------------
__global__ __launch_bounds__(256)
void cvt_transpose_kernel(const float* __restrict__ in, bf16* __restrict__ out,
                          int rows, int cols)
{
    __shared__ float tile[32][33];
    const size_t bofs = (size_t)blockIdx.z * rows * cols;
    in  += bofs;
    out += bofs;
    const int c0 = blockIdx.x * 32, r0 = blockIdx.y * 32;
    #pragma unroll
    for (int j = 0; j < 32; j += 8)
        tile[threadIdx.y + j][threadIdx.x] =
            in[(size_t)(r0 + threadIdx.y + j) * cols + c0 + threadIdx.x];
    __syncthreads();
    #pragma unroll
    for (int j = 0; j < 32; j += 8)
        out[(size_t)(c0 + threadIdx.y + j) * rows + r0 + threadIdx.x] =
            (bf16)tile[threadIdx.x][threadIdx.y + j];
}

// ---------------------------------------------------------------------------
// QKV projection: per (b,h): [T, D] x W^T[hs, D] (+bias) -> bf16.
// trans_out=0: out[z][t][s]  (q, k);  trans_out=1: out[z][s][t]  (v).
// grid (1, T/32, B*H)
// ---------------------------------------------------------------------------
__global__ __launch_bounds__(256)
void qkv_proj_kernel(const bf16* __restrict__ srcb, const bf16* __restrict__ WT,
                     const float* __restrict__ bias, bf16* __restrict__ out,
                     int trans_out)
{
    __shared__ __align__(16) bf16 lA[32 * 32];
    __shared__ __align__(16) bf16 lB[64 * 32];
    const int z  = blockIdx.z;
    const int h  = z & (NHEAD - 1);
    const int b  = z >> 4;
    const int m0 = blockIdx.y * 32;
    const bf16* A  = srcb + ((size_t)b * T_SEQ + m0) * DMODEL;
    const bf16* BT = WT + (size_t)h * HSZ * DMODEL;     // [hs, D]
    v8f acc = gemm_core_n64<false>(A, DMODEL, BT, DMODEL, DMODEL, lA, lB);

    EPI1_COORDS();
    const int n  = no_ + (lane & 15);
    const int mb = m0 + mo_;
    const float bv = bias ? bias[h * HSZ + n] : 0.f;
    bf16* orow = out + (size_t)z * T_SEQ * HSZ;
    if (trans_out) {
        #pragma unroll
        for (int i = 0; i < 8; ++i)
            orow[(size_t)n * T_SEQ + mb + i] = (bf16)(acc[i] + bv);
    } else {
        #pragma unroll
        for (int i = 0; i < 8; ++i)
            orow[(size_t)(mb + i) * HSZ + n] = (bf16)(acc[i] + bv);
    }
}

// ---------------------------------------------------------------------------
// scores = sqrt(hs) * q . k^T per (b,h).  grid (T/128, T/64, B*H)
// k is stored [T, hs] = [N, K] row-major already.
// ---------------------------------------------------------------------------
__global__ __launch_bounds__(256)
void qk_scores_kernel(const bf16* __restrict__ qb, const bf16* __restrict__ kb,
                      float* __restrict__ scores)
{
    __shared__ __align__(16) bf16 lA[64 * 64];
    __shared__ __align__(16) bf16 lB[128 * 64];
    const int z  = blockIdx.z;
    const int m0 = blockIdx.y * 64;
    const int n0 = blockIdx.x * 128;
    const v8f z8 = {0.f, 0.f, 0.f, 0.f, 0.f, 0.f, 0.f, 0.f};
    v8f acc[4] = {z8, z8, z8, z8};
    gemm_core2<false>(qb + ((size_t)z * T_SEQ + m0) * HSZ, HSZ,
                      kb + ((size_t)z * T_SEQ + n0) * HSZ, HSZ, HSZ, lA, lB, acc);

    EPI2_COORDS();
    float* o = scores + (size_t)z * T_SEQ * T_SEQ;
    #pragma unroll
    for (int sm = 0; sm < 2; ++sm)
        #pragma unroll
        for (int sn = 0; sn < 2; ++sn) {
            const v8f a = acc[sm * 2 + sn];
            const int col = n0 + wn_ + sn * 16;
            #pragma unroll
            for (int i = 0; i < 8; ++i)
                o[(size_t)(m0 + wm_ + sm * 16 + i) * T_SEQ + col] = 8.0f * a[i];
        }
}

// ---------------------------------------------------------------------------
// Rel-position bias: scores[b,h,t,v] += sum_s q[b,h,t,s] * rel[h,t,v,s]
// Per (h,t): A = q over batches [16(=8 used) x 64] bf16, B = rel[h,t] [512, 64]
// ([N,K] row-major) streamed fp32->bf16 once.  grid (T/8, H), block = 8 waves.
// ---------------------------------------------------------------------------
__global__ __launch_bounds__(256)
void rel_scores_kernel(const bf16* __restrict__ qb, const float* __restrict__ rel,
                       float* __restrict__ scores)
{
    const int h    = blockIdx.y;
    const int t    = blockIdx.x * 8 + (threadIdx.x >> 5);
    const int lane = threadIdx.x & 31;
    const int r    = lane & 15;

    const uint4 z4 = make_uint4(0u, 0u, 0u, 0u);
    Frag16 a0, a1;                      // K chunks 0..31 and 32..63 of q rows
    a0.u[0] = z4; a0.u[1] = z4; a1.u[0] = z4; a1.u[1] = z4;
    if (r < NB) {                       // rows 8..15 of A stay zero (M padding)
        const bf16* q = qb + (((size_t)r * NHEAD + h) * T_SEQ + t) * HSZ;
        const int kb = (lane < 16) ? 0 : 16;
        a0.u[0] = *(const uint4*)(q + kb);
        a0.u[1] = *(const uint4*)(q + 8 + kb);
        a1.u[0] = *(const uint4*)(q + 32 + kb);
        a1.u[1] = *(const uint4*)(q + 40 + kb);
    }

    const int n  = lane & 15;
    const int ko = (lane < 16) ? 0 : 16;
    const float* relbase = rel + ((size_t)h * RELBLK + t) * (size_t)RELBLK * HSZ;

    for (int v0 = 0; v0 < T_SEQ; v0 += 16) {
        const float* rr = relbase + (size_t)(v0 + n) * HSZ + ko;
        if (v0 + 16 < T_SEQ)
            __builtin_prefetch(relbase + (size_t)(v0 + 16 + n) * HSZ + ko, 0, 1);

        const float4* rp = (const float4*)rr;
        const float4 f0 = rp[0], f1 = rp[1], f2 = rp[2], f3 = rp[3];
        const float4 g0 = rp[8], g1 = rp[9], g2 = rp[10], g3 = rp[11];  // +32 floats
        Frag16 b0, b1;
        b0.h[0]=(bf16)f0.x; b0.h[1]=(bf16)f0.y; b0.h[2]=(bf16)f0.z; b0.h[3]=(bf16)f0.w;
        b0.h[4]=(bf16)f1.x; b0.h[5]=(bf16)f1.y; b0.h[6]=(bf16)f1.z; b0.h[7]=(bf16)f1.w;
        b0.h[8]=(bf16)f2.x; b0.h[9]=(bf16)f2.y; b0.h[10]=(bf16)f2.z; b0.h[11]=(bf16)f2.w;
        b0.h[12]=(bf16)f3.x; b0.h[13]=(bf16)f3.y; b0.h[14]=(bf16)f3.z; b0.h[15]=(bf16)f3.w;
        b1.h[0]=(bf16)g0.x; b1.h[1]=(bf16)g0.y; b1.h[2]=(bf16)g0.z; b1.h[3]=(bf16)g0.w;
        b1.h[4]=(bf16)g1.x; b1.h[5]=(bf16)g1.y; b1.h[6]=(bf16)g1.z; b1.h[7]=(bf16)g1.w;
        b1.h[8]=(bf16)g2.x; b1.h[9]=(bf16)g2.y; b1.h[10]=(bf16)g2.z; b1.h[11]=(bf16)g2.w;
        b1.h[12]=(bf16)g3.x; b1.h[13]=(bf16)g3.y; b1.h[14]=(bf16)g3.z; b1.h[15]=(bf16)g3.w;

        v8f acc = {0.f, 0.f, 0.f, 0.f, 0.f, 0.f, 0.f, 0.f};
        acc = wmma_bf16(a0, b0, acc);
        acc = wmma_bf16(a1, b1, acc);
        if (lane < 16) {                // lanes 0..15 hold C rows 0..7 (= batch)
            #pragma unroll
            for (int i = 0; i < NB; ++i) {
                const size_t idx =
                    (((size_t)i * NHEAD + h) * T_SEQ + t) * (size_t)T_SEQ + v0 + n;
                scores[idx] += acc[i];
            }
        }
    }
}

// ---------------------------------------------------------------------------
// Row softmax over last dim (512). One wave per row. grid (B*H*T/8), block 256.
// ---------------------------------------------------------------------------
__global__ __launch_bounds__(256)
void softmax_kernel(float* __restrict__ scores)
{
    const size_t row = (size_t)blockIdx.x * 8 + (threadIdx.x >> 5);
    const int lane = threadIdx.x & 31;
    float* r = scores + row * T_SEQ;
    float v[16];
    float mx = -3.4e38f;
    #pragma unroll
    for (int j = 0; j < 16; ++j) { v[j] = r[lane + 32 * j]; mx = fmaxf(mx, v[j]); }
    #pragma unroll
    for (int o = 16; o > 0; o >>= 1) mx = fmaxf(mx, __shfl_xor(mx, o));
    float s = 0.f;
    #pragma unroll
    for (int j = 0; j < 16; ++j) { v[j] = __expf(v[j] - mx); s += v[j]; }
    #pragma unroll
    for (int o = 16; o > 0; o >>= 1) s += __shfl_xor(s, o);
    const float inv = 1.f / s;
    #pragma unroll
    for (int j = 0; j < 16; ++j) r[lane + 32 * j] = v[j] * inv;
}

// ---------------------------------------------------------------------------
// attn = probs (fp32, converted at stage time) @ V (V stored transposed [hs,T]),
// written head-concatenated into [B*T, D] bf16.  grid (1, T/32, B*H)
// ---------------------------------------------------------------------------
__global__ __launch_bounds__(256)
void attn_v_kernel(const float* __restrict__ probs, const bf16* __restrict__ vbt,
                   bf16* __restrict__ attnc)
{
    __shared__ __align__(16) bf16 lA[32 * 32];
    __shared__ __align__(16) bf16 lB[64 * 32];
    const int z  = blockIdx.z;
    const int h  = z & (NHEAD - 1);
    const int b  = z >> 4;
    const int m0 = blockIdx.y * 32;
    const float* A  = probs + (size_t)z * T_SEQ * T_SEQ + (size_t)m0 * T_SEQ;
    const bf16*  BT = vbt + (size_t)z * T_SEQ * HSZ;     // [hs, T]
    v8f acc = gemm_core_n64<true>(A, T_SEQ, BT, T_SEQ, T_SEQ, lA, lB);

    EPI1_COORDS();
    const int n  = no_ + (lane & 15);
    const int mb = m0 + mo_;
    #pragma unroll
    for (int i = 0; i < 8; ++i)
        attnc[((size_t)b * T_SEQ + mb + i) * DMODEL + h * HSZ + n] = (bf16)acc[i];
}

// ---------------------------------------------------------------------------
// Dense GEMM, B pre-transposed [N,K]. EPI: 0 = fp32 out, 1 = GELU -> bf16 out.
// B panels staged by the Tensor Data Mover (double buffered) when available.
// grid (N/128, M/64)
// ---------------------------------------------------------------------------
template<int EPI>
__global__ __launch_bounds__(256)
void dense_gemm_kernel(const bf16* __restrict__ A, int lda, int K,
                       const bf16* __restrict__ BT, int ldb,
                       const float* __restrict__ bias,
                       void* __restrict__ out, int ldo)
{
    __shared__ __align__(16) bf16 lA[64 * 64];
    const int m0 = blockIdx.y * 64;
    const int n0 = blockIdx.x * 128;
    const v8f z8 = {0.f, 0.f, 0.f, 0.f, 0.f, 0.f, 0.f, 0.f};
    v8f acc[4] = {z8, z8, z8, z8};
#if USE_TDM
    __shared__ __align__(16) bf16 lB0[128 * 64];
    __shared__ __align__(16) bf16 lB1[128 * 64];
    gemm_core2_tdm(A + (size_t)m0 * lda, lda, BT + (size_t)n0 * ldb, ldb, K,
                   lA, lB0, lB1, acc);
#else
    __shared__ __align__(16) bf16 lB[128 * 64];
    gemm_core2<false>(A + (size_t)m0 * lda, lda, BT + (size_t)n0 * ldb, ldb, K,
                      lA, lB, acc);
#endif

    EPI2_COORDS();
    #pragma unroll
    for (int sm = 0; sm < 2; ++sm)
        #pragma unroll
        for (int sn = 0; sn < 2; ++sn) {
            const v8f a = acc[sm * 2 + sn];
            const int col = n0 + wn_ + sn * 16;
            const float bv = bias[col];
            #pragma unroll
            for (int i = 0; i < 8; ++i) {
                const size_t row = (size_t)(m0 + wm_ + sm * 16 + i);
                float x = a[i] + bv;
                if (EPI == 1) {
                    x = 0.5f * x * (1.f + erff(x * 0.70710678118654752f)); // exact GELU
                    ((bf16*)out)[row * ldo + col] = (bf16)x;
                } else {
                    ((float*)out)[row * ldo + col] = x;
                }
            }
        }
}

// ---------------------------------------------------------------------------
// y = LN(resid + add) * g + beta ; optional bf16 mirror. One wave per row.
// grid (B*T/8), block 256.
// ---------------------------------------------------------------------------
__global__ __launch_bounds__(256)
void add_ln_kernel(const float* __restrict__ resid, const float* __restrict__ add,
                   const float* __restrict__ g, const float* __restrict__ be,
                   float* __restrict__ out, bf16* __restrict__ outb)
{
    const int row  = blockIdx.x * 8 + (threadIdx.x >> 5);
    const int lane = threadIdx.x & 31;
    const size_t base = (size_t)row * DMODEL;
    float v[32];
    float s = 0.f, s2 = 0.f;
    #pragma unroll
    for (int j = 0; j < 32; ++j) {
        const int c = lane + 32 * j;
        const float x = resid[base + c] + add[base + c];
        v[j] = x; s += x; s2 += x * x;
    }
    #pragma unroll
    for (int o = 16; o > 0; o >>= 1) { s += __shfl_xor(s, o); s2 += __shfl_xor(s2, o); }
    const float mu  = s * (1.f / DMODEL);
    const float var = s2 * (1.f / DMODEL) - mu * mu;
    const float inv = rsqrtf(var + 1e-5f);
    #pragma unroll
    for (int j = 0; j < 32; ++j) {
        const int c = lane + 32 * j;
        const float y = (v[j] - mu) * inv * g[c] + be[c];
        out[base + c] = y;
        if (outb) outb[base + c] = (bf16)y;
    }
}

// ---------------------------------------------------------------------------
// Workspace layout (bytes, ~369 MB total)
// ---------------------------------------------------------------------------
static const size_t OFF_SRCB  = 0;                     // 8 MB  src bf16 [B*T, D]
static const size_t OFF_WQ    = 8388608;               // 2 MB  Wq^T [H, hs, D]
static const size_t OFF_WK    = 10485760;              // 2 MB  Wk^T
static const size_t OFF_WV    = 12582912;              // 2 MB  Wv^T
static const size_t OFF_WO    = 14680064;              // 2 MB  Wo^T [D, D]
static const size_t OFF_WF1   = 16777216;              // 20 MB Wf1^T [DFF, D]
static const size_t OFF_WF2   = 37748736;              // 20 MB Wf2^T [D, DFF]
static const size_t OFF_QB    = 58720256;              // 8 MB  q bf16 [B,H,T,hs]
static const size_t OFF_KB    = 67108864;              // 8 MB  k bf16 [B,H,T,hs]
static const size_t OFF_VB    = 75497472;              // 8 MB  v bf16 [B,H,hs,T]
static const size_t OFF_SCORE = 83886080;              // 128 MB fp32 [B,H,T,T]
static const size_t OFF_ATTNC = 218103808;             // 8 MB  bf16 [B*T, D]
static const size_t OFF_PROJ  = 226492416;             // 16 MB fp32
static const size_t OFF_X     = 243269632;             // 16 MB fp32
static const size_t OFF_XB    = 260046848;             // 8 MB  bf16
static const size_t OFF_H1    = 268435456;             // 80 MB bf16 [B*T, DFF]
static const size_t OFF_FF    = 352321536;             // 16 MB fp32

extern "C" void kernel_launch(void* const* d_in, const int* in_sizes, int n_in,
                              void* d_out, int out_size, void* d_ws, size_t ws_size,
                              hipStream_t stream)
{
    (void)in_sizes; (void)n_in; (void)out_size; (void)ws_size;

    const float* src = (const float*)d_in[0];
    const float* Wq  = (const float*)d_in[1];
    const float* bq  = (const float*)d_in[2];
    const float* Wk  = (const float*)d_in[3];
    const float* bk  = (const float*)d_in[4];
    const float* Wv  = (const float*)d_in[5];
    const float* rel = (const float*)d_in[6];
    const float* Wo  = (const float*)d_in[7];
    const float* bo  = (const float*)d_in[8];
    const float* g1  = (const float*)d_in[9];
    const float* be1 = (const float*)d_in[10];
    const float* g2  = (const float*)d_in[11];
    const float* be2 = (const float*)d_in[12];
    const float* Wf1 = (const float*)d_in[13];
    const float* bf1 = (const float*)d_in[14];
    const float* Wf2 = (const float*)d_in[15];
    const float* bf2 = (const float*)d_in[16];

    char* ws = (char*)d_ws;
    bf16*  srcb  = (bf16*) (ws + OFF_SRCB);
    bf16*  wqb   = (bf16*) (ws + OFF_WQ);
    bf16*  wkb   = (bf16*) (ws + OFF_WK);
    bf16*  wvb   = (bf16*) (ws + OFF_WV);
    bf16*  wob   = (bf16*) (ws + OFF_WO);
    bf16*  wf1b  = (bf16*) (ws + OFF_WF1);
    bf16*  wf2b  = (bf16*) (ws + OFF_WF2);
    bf16*  qb    = (bf16*) (ws + OFF_QB);
    bf16*  kb    = (bf16*) (ws + OFF_KB);
    bf16*  vb    = (bf16*) (ws + OFF_VB);
    float* score = (float*)(ws + OFF_SCORE);
    bf16*  attnc = (bf16*) (ws + OFF_ATTNC);
    float* proj  = (float*)(ws + OFF_PROJ);
    float* x     = (float*)(ws + OFF_X);
    bf16*  xb    = (bf16*) (ws + OFF_XB);
    bf16*  h1b   = (bf16*) (ws + OFF_H1);
    float* ff    = (float*)(ws + OFF_FF);

    const dim3 blk(256);

    // 1a. plain fp32 -> bf16 for src (A operand, stays [B*T, D])
    {
        const long n4 = ((long)NB * T_SEQ * DMODEL) >> 2;
        cvt_f32_bf16_kernel<<<dim3((unsigned)(n4 / 256)), blk, 0, stream>>>(src, srcb, n4);
    }
    // 1b. transpose + convert all GEMM B operands to [N,K] bf16
    auto cvtT = [&](const float* p, bf16* o, int rows, int cols, int batch) {
        cvt_transpose_kernel<<<dim3(cols / 32, rows / 32, batch), dim3(32, 8), 0, stream>>>(
            p, o, rows, cols);
    };
    cvtT(Wq,  wqb,  DMODEL, HSZ,    NHEAD);   // -> [H, hs, D]
    cvtT(Wk,  wkb,  DMODEL, HSZ,    NHEAD);
    cvtT(Wv,  wvb,  DMODEL, HSZ,    NHEAD);
    cvtT(Wo,  wob,  DMODEL, DMODEL, 1);       // -> [D, D] (N,K)
    cvtT(Wf1, wf1b, DMODEL, DFF,    1);       // -> [DFF, D]
    cvtT(Wf2, wf2b, DFF,    DMODEL, 1);       // -> [D, DFF]

    // 2. QKV projections (v: no bias, stored transposed [hs, T])
    const dim3 gQKV(1, T_SEQ / 32, NB * NHEAD);
    qkv_proj_kernel<<<gQKV, blk, 0, stream>>>(srcb, wqb, bq, qb, 0);
    qkv_proj_kernel<<<gQKV, blk, 0, stream>>>(srcb, wkb, bk, kb, 0);
    qkv_proj_kernel<<<gQKV, blk, 0, stream>>>(srcb, wvb, nullptr, vb, 1);

    // 3. scores = sqrt(hs) * q k^T
    qk_scores_kernel<<<dim3(T_SEQ / 128, T_SEQ / 64, NB * NHEAD), blk, 0, stream>>>(
        qb, kb, score);

    // 4. += relative-position bias (streams the 1 GB rel tensor once)
    rel_scores_kernel<<<dim3(T_SEQ / 8, NHEAD), blk, 0, stream>>>(qb, rel, score);

    // 5. softmax (in place)
    softmax_kernel<<<dim3((NB * NHEAD * T_SEQ) / 8), blk, 0, stream>>>(score);

    // 6. attn = probs @ V, head-concatenated
    attn_v_kernel<<<dim3(1, T_SEQ / 32, NB * NHEAD), blk, 0, stream>>>(score, vb, attnc);

    // 7. out-projection
    dense_gemm_kernel<0><<<dim3(DMODEL / 128, (NB * T_SEQ) / 64), blk, 0, stream>>>(
        attnc, DMODEL, DMODEL, wob, DMODEL, bo, proj, DMODEL);

    // 8. x = LN(src + out)
    add_ln_kernel<<<dim3((NB * T_SEQ) / 8), blk, 0, stream>>>(src, proj, g1, be1, x, xb);

    // 9. FFN up + exact GELU
    dense_gemm_kernel<1><<<dim3(DFF / 128, (NB * T_SEQ) / 64), blk, 0, stream>>>(
        xb, DMODEL, DMODEL, wf1b, DMODEL, bf1, h1b, DFF);

    // 10. FFN down
    dense_gemm_kernel<0><<<dim3(DMODEL / 128, (NB * T_SEQ) / 64), blk, 0, stream>>>(
        h1b, DFF, DFF, wf2b, DFF, bf2, ff, DMODEL);

    // 11. out = LN(x + ff)
    add_ln_kernel<<<dim3((NB * T_SEQ) / 8), blk, 0, stream>>>(x, ff, g2, be2,
                                                              (float*)d_out, nullptr);
}
